// RBFLoss_77764677861986
// MI455X (gfx1250) — matrix operand order repaired and checked by reading, not compiled, not run on hardware
//
#include <hip/hip_runtime.h>
#include <hip/hip_bf16.h>
#include <cstddef>

#define BSZ   8192
#define DDIM  256
#define NPOSP 131072
#define GAMMA 0.5f
#define TEMP  0.07f
#define INV_T (1.0f / 0.07f)

typedef float  v2f   __attribute__((ext_vector_type(2)));
typedef float  v8f   __attribute__((ext_vector_type(8)));
typedef __bf16 v8bf  __attribute__((ext_vector_type(8)));
typedef __bf16 v16bf __attribute__((ext_vector_type(16)));

// ---------------- Kernel A: row squared norms (one wave32 per row) ----------
__global__ __launch_bounds__(256) void rbf_sqnorm(const float* __restrict__ X,
                                                  float* __restrict__ sq) {
    const int tid  = threadIdx.x;
    const int lane = tid & 31;
    const int wave = (blockIdx.x * blockDim.x + tid) >> 5;
    if (wave >= BSZ) return;
    const float* xp = X + (size_t)wave * DDIM + lane * 8;
    float s = 0.f;
#pragma unroll
    for (int i = 0; i < 8; ++i) { float v = xp[i]; s = fmaf(v, v, s); }
#pragma unroll
    for (int m = 1; m < 32; m <<= 1) s += __shfl_xor(s, m, 32);
    if (lane == 0) sq[wave] = s;
}

// ---------------- Kernel E: fp32 -> bf16 hi/lo split ------------------------
__global__ __launch_bounds__(256) void rbf_split(const float* __restrict__ X,
                                                 __bf16* __restrict__ hi,
                                                 __bf16* __restrict__ lo) {
    const int n = BSZ * DDIM;
    for (int i = blockIdx.x * blockDim.x + threadIdx.x; i < n;
         i += gridDim.x * blockDim.x) {
        const float x = X[i];
        const __bf16 h = (__bf16)x;
        hi[i] = h;
        lo[i] = (__bf16)(x - (float)h);
    }
}

// Contiguous 16-element bf16 fragment (B matrix: K = half*16 + [0..16)).
__device__ inline v16bf ldfrag_b(const __bf16* p) {
    v8bf x0 = *(const v8bf*)(p);
    v8bf x1 = *(const v8bf*)(p + 8);
    return __builtin_shufflevector(x0, x1, 0,1,2,3,4,5,6,7,8,9,10,11,12,13,14,15);
}
// A fragment: contiguous chunks at +0 and +8 (K = half*16 + [0..16)).
__device__ inline v16bf ldfrag_a(const __bf16* p) {
    v8bf x0 = *(const v8bf*)(p);
    v8bf x1 = *(const v8bf*)(p + 8);
    return __builtin_shufflevector(x0, x1, 0,1,2,3,4,5,6,7,8,9,10,11,12,13,14,15);
}

// ---------------- Kernel B (bf16 split): Gram + RBF softmax denominator -----
// 256 workgroups; each owns a 32-row stripe and sweeps all 8192 columns.
// 8 waves = 2 row-tiles x 4 column walkers; each wave processes TWO column
// tiles per iteration with a one-K-block software pipeline on the B loads.
__global__ __launch_bounds__(256) void rbf_rowsum_bf16(
        const __bf16* __restrict__ Xhi, const __bf16* __restrict__ Xlo,
        const float* __restrict__ sq, float* __restrict__ sumexp) {
    __shared__ float Wsum[4][32];

    const int tid    = threadIdx.x;
    const int lane   = tid & 31;
    const int wave   = tid >> 5;
    const int stripe = blockIdx.x * 32;

    const int rt   = wave >> 2;      // row tile 0/1
    const int cw   = wave & 3;       // column-walk phase
    const int l16  = lane & 15;
    const int half = lane >> 4;

    // Preload loop-invariant A fragments (hi & lo) into registers: 128 VGPRs.
    const size_t arow = (size_t)(stripe + rt * 16 + l16) * DDIM;
    const int akoff = half * 16;     // contiguous K layout, matches B
    v16bf ahi[8], alo[8];
#pragma unroll
    for (int kb8 = 0; kb8 < 8; ++kb8) {
        ahi[kb8] = ldfrag_a(Xhi + arow + kb8 * 32 + akoff);
        alo[kb8] = ldfrag_a(Xlo + arow + kb8 * 32 + akoff);
    }

    float sqr[8];
#pragma unroll
    for (int r = 0; r < 8; ++r) sqr[r] = sq[stripe + rt * 16 + half * 8 + r];

    float acc[8];
#pragma unroll
    for (int r = 0; r < 8; ++r) acc[r] = 0.f;

    const int bkoff = half * 16;     // contiguous K layout, matches A

    for (int n = cw; n < BSZ / 16; n += 8) {
        const int col0 = n * 16;                      // tile 0
        const int col1 = col0 + 64;                   // tile 1 (n + 4)
        const size_t brow = (size_t)(col0 + l16) * DDIM + bkoff;
        const __bf16* p0h = Xhi + brow;
        const __bf16* p0l = Xlo + brow;
        const __bf16* p1h = p0h + 64 * DDIM;
        const __bf16* p1l = p0l + 64 * DDIM;
        const float sqc0 = sq[col0 + l16];
        const float sqc1 = sq[col1 + l16];

        v8f c0 = {0.f, 0.f, 0.f, 0.f, 0.f, 0.f, 0.f, 0.f};
        v8f c1 = {0.f, 0.f, 0.f, 0.f, 0.f, 0.f, 0.f, 0.f};

        // Software pipeline: B fragments for block kb+1 load under block kb's WMMAs.
        v16bf b0h = ldfrag_b(p0h);
        v16bf b0l = ldfrag_b(p0l);
        v16bf b1h = ldfrag_b(p1h);
        v16bf b1l = ldfrag_b(p1l);
#pragma unroll
        for (int kb8 = 0; kb8 < 8; ++kb8) {
            const int kn = (kb8 < 7) ? (kb8 * 32 + 32) : (kb8 * 32);  // clamp last
            v16bf n0h = ldfrag_b(p0h + kn);
            v16bf n0l = ldfrag_b(p0l + kn);
            v16bf n1h = ldfrag_b(p1h + kn);
            v16bf n1l = ldfrag_b(p1l + kn);

            c0 = __builtin_amdgcn_wmma_f32_16x16x32_bf16(
                     false, ahi[kb8], false, b0h, (short)0, c0, false, false);
            c1 = __builtin_amdgcn_wmma_f32_16x16x32_bf16(
                     false, ahi[kb8], false, b1h, (short)0, c1, false, false);
            c0 = __builtin_amdgcn_wmma_f32_16x16x32_bf16(
                     false, alo[kb8], false, b0h, (short)0, c0, false, false);
            c1 = __builtin_amdgcn_wmma_f32_16x16x32_bf16(
                     false, alo[kb8], false, b1h, (short)0, c1, false, false);
            c0 = __builtin_amdgcn_wmma_f32_16x16x32_bf16(
                     false, ahi[kb8], false, b0l, (short)0, c0, false, false);
            c1 = __builtin_amdgcn_wmma_f32_16x16x32_bf16(
                     false, ahi[kb8], false, b1l, (short)0, c1, false, false);

            b0h = n0h; b0l = n0l; b1h = n1h; b1l = n1l;
        }

        // C layout: vgpr r -> row M = half*8 + r, col N = l16.
#pragma unroll
        for (int r = 0; r < 8; ++r) {
            const int grow = stripe + rt * 16 + half * 8 + r;
            {
                const float d2 = fmaxf(sqr[r] + sqc0 - 2.f * c0[r], 0.f);
                const float e2 = __expf((__expf(-GAMMA * d2) - 1.f) * INV_T);
                acc[r] += (grow == col0 + l16) ? 0.f : e2;
            }
            {
                const float d2 = fmaxf(sqr[r] + sqc1 - 2.f * c1[r], 0.f);
                const float e2 = __expf((__expf(-GAMMA * d2) - 1.f) * INV_T);
                acc[r] += (grow == col1 + l16) ? 0.f : e2;
            }
        }
    }

#pragma unroll
    for (int r = 0; r < 8; ++r) {
        float s = acc[r];
#pragma unroll
        for (int m = 1; m < 16; m <<= 1) s += __shfl_xor(s, m, 32);
        acc[r] = s;
    }
    if (l16 == 0) {
#pragma unroll
        for (int r = 0; r < 8; ++r)
            Wsum[cw][rt * 16 + half * 8 + r] = acc[r];
    }
    __syncthreads();
    if (tid < 32) {   // fixed-order combine -> deterministic
        float s = ((Wsum[0][tid] + Wsum[1][tid]) + Wsum[2][tid]) + Wsum[3][tid];
        sumexp[stripe + tid] = s + 1e-8f;
    }
}

// ---------------- Kernel B (f32 WMMA fallback, ws too small for bf16) -------
__global__ __launch_bounds__(256) void rbf_rowsum_f32(
        const float* __restrict__ X, const float* __restrict__ sq,
        float* __restrict__ sumexp) {
    __shared__ float As[32][260];
    __shared__ float Wsum[4][32];

    const int tid    = threadIdx.x;
    const int lane   = tid & 31;
    const int wave   = tid >> 5;
    const int stripe = blockIdx.x * 32;

    for (int idx = tid * 4; idx < 32 * 256; idx += 256 * 4) {
        const int r = idx >> 8, c = idx & 255;
        *(float4*)(&As[r][c]) = *(const float4*)(X + (size_t)(stripe + r) * DDIM + c);
    }
    __syncthreads();

    const int rt   = wave >> 2;
    const int cw   = wave & 3;
    const int l16  = lane & 15;
    const int half = lane >> 4;
    const int koff = half * 2;

    float sqr[8];
#pragma unroll
    for (int r = 0; r < 8; ++r) sqr[r] = sq[stripe + rt * 16 + half * 8 + r];

    const float* aLds = &As[rt * 16 + l16][koff];

    float acc[8];
#pragma unroll
    for (int r = 0; r < 8; ++r) acc[r] = 0.f;

    for (int n = cw; n < BSZ / 16; n += 4) {
        const int col0 = n * 16;
        const float* bp = X + (size_t)(col0 + l16) * DDIM + koff;
        const float sqc = sq[col0 + l16];

        v8f c = {0.f, 0.f, 0.f, 0.f, 0.f, 0.f, 0.f, 0.f};
#pragma unroll
        for (int k = 0; k < DDIM; k += 4) {
            v2f a = *(const v2f*)(aLds + k);
            v2f b = *(const v2f*)(bp + k);
            c = __builtin_amdgcn_wmma_f32_16x16x4_f32(
                    false, a, false, b, (short)0, c, false, false);
        }
#pragma unroll
        for (int r = 0; r < 8; ++r) {
            const float d2 = fmaxf(sqr[r] + sqc - 2.f * c[r], 0.f);
            const float e2 = __expf((__expf(-GAMMA * d2) - 1.f) * INV_T);
            const int grow = stripe + rt * 16 + half * 8 + r;
            acc[r] += (grow == col0 + l16) ? 0.f : e2;
        }
    }

#pragma unroll
    for (int r = 0; r < 8; ++r) {
        float s = acc[r];
#pragma unroll
        for (int m = 1; m < 16; m <<= 1) s += __shfl_xor(s, m, 32);
        acc[r] = s;
    }
    if (l16 == 0) {
#pragma unroll
        for (int r = 0; r < 8; ++r)
            Wsum[cw][rt * 16 + half * 8 + r] = acc[r];
    }
    __syncthreads();
    if (tid < 32) {
        float s = ((Wsum[0][tid] + Wsum[1][tid]) + Wsum[2][tid]) + Wsum[3][tid];
        sumexp[stripe + tid] = s + 1e-8f;
    }
}

// ---------------- Kernel C: positive-pair log-probs (one wave per pair) -----
__global__ __launch_bounds__(256) void rbf_pairs(const float* __restrict__ X,
                                                 const float* __restrict__ sq,
                                                 const float* __restrict__ sumexp,
                                                 const int* __restrict__ row,
                                                 const int* __restrict__ col,
                                                 float* __restrict__ partials) {
    const int tid    = threadIdx.x;
    const int lane   = tid & 31;
    const int gw     = (blockIdx.x * blockDim.x + tid) >> 5;
    const int nwaves = (gridDim.x * blockDim.x) >> 5;

    float acc = 0.f;
    for (int p = gw; p < NPOSP; p += nwaves) {
        const int r = row[p], c = col[p];
        const float* xr = X + (size_t)r * DDIM + lane * 8;
        const float* xc = X + (size_t)c * DDIM + lane * 8;
        float dot = 0.f;
#pragma unroll
        for (int i = 0; i < 8; ++i) dot = fmaf(xr[i], xc[i], dot);
#pragma unroll
        for (int m = 1; m < 32; m <<= 1) dot += __shfl_xor(dot, m, 32);
        if (lane == 0) {
            const float d2  = fmaxf(sq[r] + sq[c] - 2.f * dot, 0.f);
            const float sim = __expf(-GAMMA * d2) * INV_T;
            acc += (sim - INV_T) - __logf(sumexp[r]);   // sumexp includes +1e-8
        }
    }
    if (lane == 0) partials[gw] = acc;
}

// ---------------- Kernel D: deterministic final reduction -------------------
__global__ __launch_bounds__(256) void rbf_final(const float* __restrict__ partials,
                                                 int n, float* __restrict__ out) {
    __shared__ float red[256];
    float s = 0.f;
    for (int i = threadIdx.x; i < n; i += 256) s += partials[i];
    red[threadIdx.x] = s;
    __syncthreads();
    for (int off = 128; off > 0; off >>= 1) {
        if (threadIdx.x < off) red[threadIdx.x] += red[threadIdx.x + off];
        __syncthreads();
    }
    if (threadIdx.x == 0) out[0] = -(red[0] / (float)NPOSP) * TEMP;
}

extern "C" void kernel_launch(void* const* d_in, const int* in_sizes, int n_in,
                              void* d_out, int out_size, void* d_ws, size_t ws_size,
                              hipStream_t stream) {
    const float* X   = (const float*)d_in[0];
    const int*   row = (const int*)d_in[1];
    const int*   col = (const int*)d_in[2];
    float* out = (float*)d_out;

    float* ws       = (float*)d_ws;
    float* sq       = ws;             // 8192 floats
    float* sumexp   = ws + BSZ;       // 8192 floats
    float* partials = ws + 2 * BSZ;   // 2048 floats

    const size_t base_bytes = (size_t)(2 * BSZ + 2048) * sizeof(float);  // 73728
    const size_t elems      = (size_t)BSZ * DDIM;
    const size_t need_bf16  = base_bytes + 2 * elems * sizeof(__bf16);   // ~8.1 MB

    rbf_sqnorm<<<BSZ / 8, 256, 0, stream>>>(X, sq);

    if (ws_size >= need_bf16) {
        __bf16* Xhi = (__bf16*)((char*)d_ws + base_bytes);
        __bf16* Xlo = Xhi + elems;
        rbf_split<<<1024, 256, 0, stream>>>(X, Xhi, Xlo);
        rbf_rowsum_bf16<<<BSZ / 32, 256, 0, stream>>>(Xhi, Xlo, sq, sumexp);
    } else {
        rbf_rowsum_f32<<<BSZ / 32, 256, 0, stream>>>(X, sq, sumexp);
    }

    rbf_pairs<<<256, 256, 0, stream>>>(X, sq, sumexp, row, col, partials);
    rbf_final<<<1, 256, 0, stream>>>(partials, 2048, out);
}
